// GNN_Net_Graph_Transformer_35442070126617
// MI455X (gfx1250) — compile-verified, matrix operand order checked by reading, not run on hardware
//
#include <hip/hip_runtime.h>
#include <hip/hip_bf16.h>

// ---------------- types for WMMA ----------------
typedef __attribute__((ext_vector_type(16))) __bf16 v16bf;
typedef __attribute__((ext_vector_type(8)))  float  v8f;

union V16U { uint4 q[2]; v16bf v; };

// ---------------- helpers ----------------
__device__ __forceinline__ unsigned short f2bf(float f) {
    unsigned u = __float_as_uint(f);
    unsigned r = u + 0x7FFFu + ((u >> 16) & 1u);   // round-to-nearest-even
    return (unsigned short)(r >> 16);
}
__device__ __forceinline__ unsigned encf(float f) {   // monotonic float->uint
    unsigned u = __float_as_uint(f);
    return (u >> 31) ? ~u : (u | 0x80000000u);
}
__device__ __forceinline__ float decf(unsigned v) {
    unsigned u = (v >> 31) ? (v & 0x7FFFFFFFu) : ~v;
    return __uint_as_float(u);
}

// ---------------- elementwise / fill ----------------
__global__ void k_fill(float* p, float v, size_t n) {
    size_t i = (size_t)blockIdx.x * blockDim.x + threadIdx.x;
    size_t st = (size_t)gridDim.x * blockDim.x;
    for (; i < n; i += st) p[i] = v;
}
__global__ void k_f2bf(const float* __restrict__ in, unsigned short* __restrict__ out, size_t n) {
    size_t i = (size_t)blockIdx.x * blockDim.x + threadIdx.x;
    size_t st = (size_t)gridDim.x * blockDim.x;
    for (; i < n; i += st) out[i] = f2bf(in[i]);
}
__global__ void k_add_bf16(const float* __restrict__ a, const float* __restrict__ b,
                           unsigned short* __restrict__ out, size_t n) {
    size_t i = (size_t)blockIdx.x * blockDim.x + threadIdx.x;
    size_t st = (size_t)gridDim.x * blockDim.x;
    for (; i < n; i += st) out[i] = f2bf(a[i] + b[i]);
}

// ---------------- weight packing into WMMA B-operand lane layout ----------------
// packed index: (((nt*Ktiles + kt)*32 + lane)*16 + j)
// lane = n_local + 16*g ; element j -> k = kt*32 + (j&7) + 8*g + 16*(j>>3)
__global__ void k_pack_w(const float* __restrict__ W, unsigned short* __restrict__ out,
                         int K, int Ncols) {
    int Ktiles = K >> 5;
    int total = (Ncols >> 4) * Ktiles * 512;
    for (int idx = blockIdx.x * blockDim.x + threadIdx.x; idx < total;
         idx += gridDim.x * blockDim.x) {
        int j    = idx & 15;
        int lane = (idx >> 4) & 31;
        int kt   = (idx >> 9) % Ktiles;
        int nt   = (idx >> 9) / Ktiles;
        int n = nt * 16 + (lane & 15);
        int g = lane >> 4;
        int k = kt * 32 + (j & 7) + 8 * g + 16 * (j >> 3);
        out[idx] = f2bf(W[(size_t)k * Ncols + n]);
    }
}

// ---------------- WMMA GEMM:  C[M,Ncols] = bf16(A[M,K]) @ packedB + bias ----------------
// 4 waves per block, each wave computes one 16-row M-tile x NT 16-col N-tiles.
// Per K-step: load A fragment + ALL NT B fragments (10 b128 loads in flight),
// then issue NT back-to-back wmma ops so loads overlap matrix math.
// Out-of-range rows: loads clamped to row M-1 (valid memory), stores guarded.
// act: 0=none 1=relu 2=leaky(0.01)
template <int NT>
__global__ __launch_bounds__(128)
void k_gemm_bf16(const unsigned short* __restrict__ A,
                 const unsigned short* __restrict__ Bp,
                 const float* __restrict__ bias,
                 float* __restrict__ C,
                 int M, int K, int Ncols, int act) {
    const int lane = threadIdx.x & 31;
    const int wave = threadIdx.x >> 5;
    const int mt  = blockIdx.x * 4 + wave;
    const int nt0 = blockIdx.y * NT;
    const int g  = lane >> 4;
    const int ml = lane & 15;
    const int row  = mt * 16 + ml;
    const int lrow = row < M ? row : (M - 1);       // clamp: keep loads in-bounds
    const int Ktiles = K >> 5;

    const unsigned short* ap = A + (size_t)lrow * K + 8 * g;          // +32 elems / kt
    const unsigned short* bp0 = Bp + ((size_t)nt0 * Ktiles * 32 + lane) * 16; // +512 / kt

    v8f acc[NT];
#pragma unroll
    for (int i = 0; i < NT; ++i) acc[i] = (v8f){0.f,0.f,0.f,0.f,0.f,0.f,0.f,0.f};

    for (int kt = 0; kt < Ktiles; ++kt) {
        V16U a, b[NT];
        a.q[0] = *reinterpret_cast<const uint4*>(ap + kt * 32);
        a.q[1] = *reinterpret_cast<const uint4*>(ap + kt * 32 + 16);
#pragma unroll
        for (int i = 0; i < NT; ++i) {
            const unsigned short* bp = bp0 + ((size_t)i * Ktiles + kt) * 512;
            b[i].q[0] = *reinterpret_cast<const uint4*>(bp);
            b[i].q[1] = *reinterpret_cast<const uint4*>(bp + 16);
        }
#pragma unroll
        for (int i = 0; i < NT; ++i) {
            acc[i] = __builtin_amdgcn_wmma_f32_16x16x32_bf16(false, a.v, false, b[i].v,
                                                             (short)0, acc[i], false, false);
        }
    }

#pragma unroll
    for (int i = 0; i < NT; ++i) {
        const int n = (nt0 + i) * 16 + ml;
        const float bv = bias ? bias[n] : 0.f;
#pragma unroll
        for (int r = 0; r < 8; ++r) {
            int m = mt * 16 + r + 8 * g;
            if (m < M) {
                float v = acc[i][r] + bv;
                if (act == 1)      v = v > 0.f ? v : 0.f;
                else if (act == 2) v = v > 0.f ? v : 0.01f * v;
                C[(size_t)m * Ncols + n] = v;
            }
        }
    }
}

// ---------------- edge kernels (H == 256, wave-per-edge, 8 floats/lane) ----------------
__global__ void k_scores(const float* __restrict__ q, const float* __restrict__ kf,
                         const float* __restrict__ ef,
                         const int* __restrict__ src, const int* __restrict__ dst,
                         float* __restrict__ score, unsigned* __restrict__ mmax, int E) {
    int wid = (int)(((size_t)blockIdx.x * blockDim.x + threadIdx.x) >> 5);
    int lane = threadIdx.x & 31;
    if (wid >= E) return;
    int s = src[wid], d = dst[wid];
    const float* qp = q  + (size_t)d * 256 + lane * 8;
    const float* kp = kf + (size_t)s * 256 + lane * 8;
    const float* ep = ef + (size_t)wid * 256 + lane * 8;
    float acc = 0.f;
#pragma unroll
    for (int i = 0; i < 8; ++i) acc += qp[i] * (kp[i] + ep[i]);
    for (int off = 16; off; off >>= 1) acc += __shfl_xor(acc, off, 32);
    if (lane == 0) {
        float sc = acc * 0.0625f;           // 1/sqrt(256)
        score[wid] = sc;
        atomicMax(mmax + d, encf(sc));
    }
}
__global__ void k_expsum(const float* __restrict__ score, const int* __restrict__ dst,
                         const unsigned* __restrict__ mmax,
                         float* __restrict__ ex, float* __restrict__ denom, int E) {
    int e = blockIdx.x * blockDim.x + threadIdx.x;
    if (e >= E) return;
    int d = dst[e];
    float x = __expf(score[e] - decf(mmax[d]));
    ex[e] = x;
    atomicAdd(denom + d, x);
}
__global__ void k_aggregate(const float* __restrict__ vf, const float* __restrict__ ef,
                            const float* __restrict__ ex, const float* __restrict__ denom,
                            const int* __restrict__ src, const int* __restrict__ dst,
                            float* __restrict__ out, int E) {
    int wid = (int)(((size_t)blockIdx.x * blockDim.x + threadIdx.x) >> 5);
    int lane = threadIdx.x & 31;
    if (wid >= E) return;
    int s = src[wid], d = dst[wid];
    float alpha = ex[wid] / (denom[d] + 1e-16f);
    const float* vp = vf + (size_t)s * 256 + lane * 8;
    const float* ep = ef + (size_t)wid * 256 + lane * 8;
    float* op = out + (size_t)d * 256 + lane * 8;
#pragma unroll
    for (int i = 0; i < 8; ++i) atomicAdd(op + i, alpha * (vp[i] + ep[i]));
}
__global__ void k_gather(const float* __restrict__ h, const int* __restrict__ src,
                         const int* __restrict__ dst, float* __restrict__ agg, int E) {
    int wid = (int)(((size_t)blockIdx.x * blockDim.x + threadIdx.x) >> 5);
    int lane = threadIdx.x & 31;
    if (wid >= E) return;
    int s = src[wid], d = dst[wid];
    const float* hp = h + (size_t)s * 256 + lane * 8;
    float* ap = agg + (size_t)d * 256 + lane * 8;
#pragma unroll
    for (int i = 0; i < 8; ++i) atomicAdd(ap + i, hp[i]);
}

// ---------------- leaky-relu + layernorm (wave-per-row), writes f32 + bf16 ----------------
__global__ void k_lrelu_ln(const float* __restrict__ in, const float* __restrict__ gw,
                           const float* __restrict__ bw,
                           float* __restrict__ outf, unsigned short* __restrict__ outb, int N) {
    int wid = (int)(((size_t)blockIdx.x * blockDim.x + threadIdx.x) >> 5);
    int lane = threadIdx.x & 31;
    if (wid >= N) return;
    const float* p = in + (size_t)wid * 256 + lane * 8;
    float v[8]; float s = 0.f, s2 = 0.f;
#pragma unroll
    for (int i = 0; i < 8; ++i) {
        float t = p[i]; t = t > 0.f ? t : 0.01f * t;
        v[i] = t; s += t; s2 += t * t;
    }
    for (int off = 16; off; off >>= 1) { s += __shfl_xor(s, off, 32); s2 += __shfl_xor(s2, off, 32); }
    float mu = s * (1.f / 256.f);
    float var = s2 * (1.f / 256.f) - mu * mu;
    float rs = rsqrtf(var + 1e-5f);
    int h0 = lane * 8;
#pragma unroll
    for (int i = 0; i < 8; ++i) {
        float t = (v[i] - mu) * rs * gw[h0 + i] + bw[h0 + i];
        outf[(size_t)wid * 256 + h0 + i] = t;
        outb[(size_t)wid * 256 + h0 + i] = f2bf(t);
    }
}

// ---------------- batch-norm over axis 0 ----------------
__global__ void k_bn_stats(const float* __restrict__ y, float* __restrict__ s,
                           float* __restrict__ s2, int N) {
    int col = threadIdx.x;          // blockDim.x == 256
    int r0 = blockIdx.x * 128;
    float a = 0.f, b = 0.f;
    for (int r = 0; r < 128; ++r) {
        int row = r0 + r;
        if (row < N) { float v = y[(size_t)row * 256 + col]; a += v; b += v * v; }
    }
    atomicAdd(&s[col], a);
    atomicAdd(&s2[col], b);
}
__global__ void k_bn_apply(const float* __restrict__ y, const float* __restrict__ s,
                           const float* __restrict__ s2, const float* __restrict__ g,
                           const float* __restrict__ b, unsigned short* __restrict__ out,
                           int N) {
    size_t n = (size_t)N * 256;
    size_t i = (size_t)blockIdx.x * blockDim.x + threadIdx.x;
    size_t st = (size_t)gridDim.x * blockDim.x;
    float inv = 1.f / (float)N;
    for (; i < n; i += st) {
        int col = (int)(i & 255);
        float mu = s[col] * inv;
        float var = s2[col] * inv - mu * mu;
        float v = (y[i] - mu) * rsqrtf(var + 1e-5f) * g[col] + b[col];
        v = v > 0.f ? v : 0.f;      // ReLU inside GIN MLP
        out[i] = f2bf(v);
    }
}

// ---------------- global add pool ----------------
__global__ void k_pool(const float* __restrict__ h, const int* __restrict__ batch,
                       float* __restrict__ pooled, int N) {
    size_t n = (size_t)N * 256;
    size_t i = (size_t)blockIdx.x * blockDim.x + threadIdx.x;
    size_t st = (size_t)gridDim.x * blockDim.x;
    for (; i < n; i += st) {
        int node = (int)(i >> 8);
        int col = (int)(i & 255);
        atomicAdd(&pooled[(size_t)batch[node] * 256 + col], h[i]);
    }
}

// =====================================================================
extern "C" void kernel_launch(void* const* d_in, const int* in_sizes, int n_in,
                              void* d_out, int out_size, void* d_ws, size_t ws_size,
                              hipStream_t stream) {
    const int IN = 128, H = 256, ED = 32, OUT = 16;
    const int N = in_sizes[0] / IN;
    const int E = in_sizes[1] / 2;
    const int G = out_size / OUT;

    const float* x    = (const float*)d_in[0];
    const int*   ei   = (const int*)d_in[1];
    const int*   srcI = ei;
    const int*   dstI = ei + E;
    const float* ea   = (const float*)d_in[2];
    const int*   batch = (const int*)d_in[3];
    auto P = [&](int i) { return (const float*)d_in[i]; };

    // ---- workspace bump allocation ----
    char* wp = (char*)d_ws;
    auto alloc = [&](size_t bytes) -> char* {
        char* p = wp; wp += (bytes + 255) & ~(size_t)255; return p;
    };
    unsigned short* Abf   = (unsigned short*)alloc((size_t)N * H * 2);
    unsigned short* eabf  = (unsigned short*)alloc((size_t)E * ED * 2);
    unsigned short* packW = (unsigned short*)alloc((size_t)H * H * 2);
    float* qf    = (float*)alloc((size_t)N * H * 4);   // also GIN agg
    float* kf    = (float*)alloc((size_t)N * H * 4);   // also GIN y
    float* vf    = (float*)alloc((size_t)N * H * 4);   // also GIN1 output
    float* outf  = (float*)alloc((size_t)N * H * 4);
    float* hf    = (float*)alloc((size_t)N * H * 4);
    float* ef    = (float*)alloc((size_t)E * H * 4);
    float* score = (float*)alloc((size_t)E * 4);
    float* exb   = (float*)alloc((size_t)E * 4);
    unsigned* mmax = (unsigned*)alloc((size_t)N * 4);
    float* denom = (float*)alloc((size_t)N * 4);
    float* bns   = (float*)alloc(H * 4);
    float* bns2  = (float*)alloc(H * 4);
    float* pooled = (float*)alloc((size_t)G * H * 4);
    float* zf     = (float*)alloc((size_t)G * H * 4);

    const int TB = 256;
    auto nb = [&](size_t n) { return (unsigned)((n + TB - 1) / TB); };
    const unsigned EW = nb((size_t)E * 32);   // wave-per-edge grids
    const unsigned NW = nb((size_t)N * 32);

    auto gemm = [&](const unsigned short* A, const float* W, const float* bias, float* C,
                    int M, int K, int Ncols, int act) {
        int Kt = K / 32, Nt = Ncols / 16;
        int total = Nt * Kt * 512;
        k_pack_w<<<nb(total), TB, 0, stream>>>(W, packW, K, Ncols);
        if ((Ncols & 63) == 0) {
            dim3 gr((unsigned)((M + 63) / 64), (unsigned)(Ncols / 64));
            k_gemm_bf16<4><<<gr, 128, 0, stream>>>(A, packW, bias, C, M, K, Ncols, act);
        } else {
            dim3 gr((unsigned)((M + 63) / 64), (unsigned)Nt);
            k_gemm_bf16<1><<<gr, 128, 0, stream>>>(A, packW, bias, C, M, K, Ncols, act);
        }
    };

    // ---- one-time converts ----
    k_f2bf<<<4096, TB, 0, stream>>>(x, Abf, (size_t)N * IN);
    k_f2bf<<<4096, TB, 0, stream>>>(ea, eabf, (size_t)E * ED);

    auto tconv = [&](int din, int pi) {     // pi = index of Wq
        gemm(Abf, P(pi + 0), P(pi + 1), qf, N, din, H, 0);     // q = x@Wq+bq
        gemm(Abf, P(pi + 2), P(pi + 3), kf, N, din, H, 0);     // k
        gemm(Abf, P(pi + 4), P(pi + 5), vf, N, din, H, 0);     // v
        gemm(eabf, P(pi + 6), nullptr,   ef, E, ED,  H, 0);    // e = ea@We
        gemm(Abf, P(pi + 7), P(pi + 8), outf, N, din, H, 0);   // skip: x@Ws+bs
        k_fill<<<nb(N), TB, 0, stream>>>((float*)mmax, 0.f, (size_t)N);
        k_fill<<<nb(N), TB, 0, stream>>>(denom, 0.f, (size_t)N);
        k_scores<<<EW, TB, 0, stream>>>(qf, kf, ef, srcI, dstI, score, mmax, E);
        k_expsum<<<nb(E), TB, 0, stream>>>(score, dstI, mmax, exb, denom, E);
        k_aggregate<<<EW, TB, 0, stream>>>(vf, ef, exb, denom, srcI, dstI, outf, E);
    };

    // ---- conv1 + LN1 ----
    tconv(IN, 4);
    k_lrelu_ln<<<NW, TB, 0, stream>>>(outf, P(13), P(14), hf, Abf, N);
    // ---- conv2 + LN2 ----
    tconv(H, 15);
    k_lrelu_ln<<<NW, TB, 0, stream>>>(outf, P(24), P(25), hf, Abf, N);

    // ---- GIN convs ----
    auto gin = [&](const float* hin, int pi, int actOut, float* hout) {
        float* agg = qf;
        float* yf  = kf;
        k_fill<<<4096, TB, 0, stream>>>(agg, 0.f, (size_t)N * H);
        k_gather<<<EW, TB, 0, stream>>>(hin, srcI, dstI, agg, E);
        k_add_bf16<<<4096, TB, 0, stream>>>(hin, agg, Abf, (size_t)N * H);
        gemm(Abf, P(pi + 0), P(pi + 1), yf, N, H, H, 0);       // y = h2@W1+b1
        k_fill<<<nb(H), TB, 0, stream>>>(bns, 0.f, (size_t)H);
        k_fill<<<nb(H), TB, 0, stream>>>(bns2, 0.f, (size_t)H);
        k_bn_stats<<<(unsigned)((N + 127) / 128), H, 0, stream>>>(yf, bns, bns2, N);
        k_bn_apply<<<4096, TB, 0, stream>>>(yf, bns, bns2, P(pi + 2), P(pi + 3), Abf, N);
        gemm(Abf, P(pi + 4), P(pi + 5), hout, N, H, H, actOut); // @W2+b2 (+outer relu)
    };
    gin(hf, 26, /*relu between gins*/1, vf);
    gin(vf, 32, 0, hf);

    // ---- pool + head ----
    k_fill<<<nb((size_t)G * H), TB, 0, stream>>>(pooled, 0.f, (size_t)G * H);
    k_pool<<<4096, TB, 0, stream>>>(hf, batch, pooled, N);
    k_f2bf<<<nb((size_t)G * H), TB, 0, stream>>>(pooled, Abf, (size_t)G * H);
    gemm(Abf, P(38), P(39), zf, G, H, H, 1);                  // relu(pooled@lin+b)
    k_f2bf<<<nb((size_t)G * H), TB, 0, stream>>>(zf, Abf, (size_t)G * H);
    gemm(Abf, P(40), P(41), (float*)d_out, G, H, OUT, 0);     // logits
}
